// ContrastiveLoss_25778393710638
// MI455X (gfx1250) — matrix operand order repaired and checked by reading, not compiled, not run on hardware
//
#include <hip/hip_runtime.h>
#include <hip/hip_bf16.h>

#define B_ROWS 16384
#define C_REAL 500
#define C_PAD  512
#define DDIM   512
#define MT 128
#define NT 128
#define KC 64
#define LDA 72          // LDS row stride in halves (64 + 8 pad -> 144B, 16B aligned)
#define MARGIN_F 1.0f

typedef __attribute__((ext_vector_type(16))) __bf16 v16bf;
typedef __attribute__((ext_vector_type(8)))  float  v8f;

union FragU { uint4 q[2]; v16bf v; };

__device__ __forceinline__ unsigned short f2bf(float f) {
  unsigned u = __float_as_uint(f);
  u += 0x7FFFu + ((u >> 16) & 1u);     // round-to-nearest-even
  return (unsigned short)(u >> 16);
}
// monotonic float<->uint encoding so atomicMin(u32) == float min (handles negatives)
__device__ __forceinline__ unsigned fenc(float f) {
  unsigned u = __float_as_uint(f);
  return (u & 0x80000000u) ? ~u : (u | 0x80000000u);
}
__device__ __forceinline__ float fdec(unsigned u) {
  unsigned b = (u & 0x80000000u) ? (u & 0x7FFFFFFFu) : ~u;
  return __uint_as_float(b);
}

__global__ void init_k(unsigned* __restrict__ negmin, float* __restrict__ out) {
  int i = blockIdx.x * blockDim.x + threadIdx.x;
  if (i < B_ROWS) negmin[i] = 0xFFFFFFFFu;   // encoded +max
  if (i == 0) out[0] = 0.0f;
}

// prototypes fp32 [500][512] -> bf16 [512][512] (zero-padded), p2[512] (pad = 1e30)
__global__ void proto_k(const float* __restrict__ protos,
                        unsigned short* __restrict__ pb,
                        float* __restrict__ p2) {
  __shared__ float red[128];
  const int c = blockIdx.x;      // 0..511
  const int t = threadIdx.x;     // 0..127, 4 floats each
  float4 v = make_float4(0.f, 0.f, 0.f, 0.f);
  if (c < C_REAL) v = ((const float4*)(protos + (size_t)c * DDIM))[t];
  ushort4 h;
  h.x = f2bf(v.x); h.y = f2bf(v.y); h.z = f2bf(v.z); h.w = f2bf(v.w);
  ((ushort4*)(pb + (size_t)c * DDIM))[t] = h;
  red[t] = v.x * v.x + v.y * v.y + v.z * v.z + v.w * v.w;
  __syncthreads();
  for (int off = 64; off > 0; off >>= 1) {
    if (t < off) red[t] += red[t + off];
    __syncthreads();
  }
  if (t == 0) p2[c] = (c < C_REAL) ? red[0] : 1e30f;
}

__global__ __launch_bounds__(256) void gemm_k(
    const float* __restrict__ feats, const unsigned short* __restrict__ pb,
    const float* __restrict__ p2, const int* __restrict__ labels,
    float* __restrict__ f2, unsigned* __restrict__ negmin,
    float* __restrict__ sqpos) {
  __shared__ unsigned char smem[65536];
  unsigned short* As = (unsigned short*)smem;   // [128][LDA]
  unsigned short* Bs = As + MT * LDA;           // [128][LDA]   (36864B total)
  float* Sq = (float*)smem;                     // [128][128] f32, aliased after GEMM

  const int t    = threadIdx.x;
  const int lane = t & 31;
  const int wave = t >> 5;
  const int wm   = wave >> 1;          // 0..3 -> M base wm*32
  const int wn   = wave & 1;           // 0..1 -> N base wn*64
  const int hi   = (lane >= 16) ? 1 : 0;
  const int l15  = lane & 15;

  const int r    = t >> 1;             // 0..127 (row owned for load/scan)
  const int half = t & 1;

  const int bm = blockIdx.y;           // 0..127
  const int bn = blockIdx.x;           // 0..3

  const v8f vzero = {0.f, 0.f, 0.f, 0.f, 0.f, 0.f, 0.f, 0.f};
  v8f acc[2][4];
#pragma unroll
  for (int mi = 0; mi < 2; ++mi)
#pragma unroll
    for (int ni = 0; ni < 4; ++ni) acc[mi][ni] = vzero;

  float f2acc = 0.f;
  const float* arow = feats + ((size_t)(bm * MT + r)) * DDIM + half * 32;
  const uint4* brow = (const uint4*)(pb + ((size_t)(bn * NT + r)) * DDIM + half * 32);

  for (int k0 = 0; k0 < DDIM; k0 += KC) {
    if (k0 + KC < DDIM) __builtin_prefetch(arow + k0 + KC, 0, 1);
    // ---- A chunk: 128x64 fp32 -> bf16 in LDS; fold in row-norm accumulation
#pragma unroll
    for (int j = 0; j < 8; ++j) {
      float4 v = ((const float4*)(arow + k0))[j];
      ushort4 h;
      h.x = f2bf(v.x); h.y = f2bf(v.y); h.z = f2bf(v.z); h.w = f2bf(v.w);
      *(ushort4*)&As[r * LDA + half * 32 + j * 4] = h;
      f2acc += v.x * v.x + v.y * v.y + v.z * v.z + v.w * v.w;
    }
    // ---- B chunk: 128x64 bf16 copy
#pragma unroll
    for (int j = 0; j < 4; ++j) {
      uint4 q = brow[(k0 >> 3) + j];
      *(uint4*)&Bs[r * LDA + half * 32 + j * 8] = q;
    }
    __syncthreads();

#pragma unroll
    for (int ks = 0; ks < KC; ks += 32) {
      FragU fa[2], fb[4];
      // A 16x32 bf16 layout: lanes<16 rows 0..15 hold K 0..7 & 16..23; lanes>=16 hold K 8..15 & 24..31
#pragma unroll
      for (int mi = 0; mi < 2; ++mi) {
        const unsigned short* ap = &As[(wm * 32 + mi * 16 + l15) * LDA + ks + hi * 8];
        fa[mi].q[0] = *(const uint4*)ap;
        fa[mi].q[1] = *(const uint4*)(ap + 16);
      }
      // B 32x16 bf16 layout: lanes<16 = cols, K 0..7 & 16..23; lanes>=16 = cols, K 8..15 & 24..31
#pragma unroll
      for (int ni = 0; ni < 4; ++ni) {
        const unsigned short* bp = &Bs[(wn * 64 + ni * 16 + l15) * LDA + ks + hi * 8];
        fb[ni].q[0] = *(const uint4*)bp;
        fb[ni].q[1] = *(const uint4*)(bp + 16);
      }
#pragma unroll
      for (int mi = 0; mi < 2; ++mi)
#pragma unroll
        for (int ni = 0; ni < 4; ++ni)
          acc[mi][ni] = __builtin_amdgcn_wmma_f32_16x16x32_bf16(
              false, fa[mi].v, false, fb[ni].v, (short)0, acc[mi][ni], false, false);
    }
    __syncthreads();
  }

  // row squared-norms (only bn==0 stores; every block computed them for free)
  float f2row = f2acc + __shfl_xor(f2acc, 1, 32);
  if (bn == 0 && half == 0) f2[bm * MT + r] = f2row;

  // ---- epilogue: s = p2[n] - 2*dot into LDS tile (f2 is a per-row constant; add later)
  float p2v[4];
#pragma unroll
  for (int ni = 0; ni < 4; ++ni) p2v[ni] = p2[bn * NT + wn * 64 + ni * 16 + l15];
#pragma unroll
  for (int mi = 0; mi < 2; ++mi) {
    const int mbase = wm * 32 + mi * 16 + hi * 8;   // C-layout: lanes>=16 hold M=8..15
#pragma unroll
    for (int ni = 0; ni < 4; ++ni) {
      const int nl = wn * 64 + ni * 16 + l15;
#pragma unroll
      for (int v = 0; v < 8; ++v)
        Sq[(mbase + v) * 128 + nl] = p2v[ni] - 2.0f * acc[mi][ni][v];
    }
  }
  __syncthreads();

  // ---- per-row scan of 128 cols (2 threads/row), rotated start to avoid bank conflicts
  const int grow   = bm * MT + r;
  const int labloc = labels[grow] - bn * NT;    // label column if it lives in this tile
  float mn = 3.0e38f;
  float pos = 0.f;
  int hasPos = 0;
#pragma unroll 4
  for (int j = 0; j < 64; ++j) {
    const int nl = half * 64 + ((j + r) & 63);
    const float s = Sq[r * 128 + nl];
    if (nl == labloc) { pos = s; hasPos = 1; }
    else mn = fminf(mn, s);                     // pad cols carry p2=1e30 -> never min
  }
  mn = fminf(mn, __shfl_xor(mn, 1, 32));
  if (hasPos) sqpos[grow] = pos;                // exactly one writer per row
  if (half == 0) atomicMin(negmin + grow, fenc(mn));
}

__global__ void loss_k(const float* __restrict__ f2, const float* __restrict__ sqpos,
                       const unsigned* __restrict__ negmin, float* __restrict__ out) {
  __shared__ float red[8];
  const int i = blockIdx.x * 256 + threadIdx.x;
  const float f2v = f2[i];
  const float dp = sqrtf(fmaxf(f2v + sqpos[i], 1e-12f));
  const float dn = sqrtf(fmaxf(f2v + fdec(negmin[i]), 1e-12f));
  float term = fmaxf(dp - dn + MARGIN_F, 0.0f);
#pragma unroll
  for (int off = 16; off > 0; off >>= 1) term += __shfl_down(term, off, 32);
  const int lane = threadIdx.x & 31, wv = threadIdx.x >> 5;
  if (lane == 0) red[wv] = term;
  __syncthreads();
  if (threadIdx.x == 0) {
    float s = 0.f;
#pragma unroll
    for (int w = 0; w < 8; ++w) s += red[w];
    atomicAdd(out, s * (1.0f / (float)B_ROWS));
  }
}

extern "C" void kernel_launch(void* const* d_in, const int* in_sizes, int n_in,
                              void* d_out, int out_size, void* d_ws, size_t ws_size,
                              hipStream_t stream) {
  (void)in_sizes; (void)n_in; (void)out_size; (void)ws_size;
  const float* feats  = (const float*)d_in[0];
  const float* protos = (const float*)d_in[1];
  const int*   labels = (const int*)d_in[2];
  float* out = (float*)d_out;

  char* w = (char*)d_ws;
  unsigned short* pb     = (unsigned short*)w;            // 512*512*2 = 524288 B
  float*          p2     = (float*)(w + 524288);          //   2048 B
  float*          f2     = (float*)(w + 526336);          //  65536 B
  unsigned*       negmin = (unsigned*)(w + 591872);       //  65536 B
  float*          sqpos  = (float*)(w + 657408);          //  65536 B (total ~707 KB)

  hipLaunchKernelGGL(init_k, dim3(64), dim3(256), 0, stream, negmin, out);
  hipLaunchKernelGGL(proto_k, dim3(C_PAD), dim3(128), 0, stream, protos, pb, p2);
  hipLaunchKernelGGL(gemm_k, dim3(C_PAD / NT, B_ROWS / MT), dim3(256), 0, stream,
                     feats, pb, p2, labels, f2, negmin, sqpos);
  hipLaunchKernelGGL(loss_k, dim3(B_ROWS / 256), dim3(256), 0, stream,
                     f2, sqpos, negmin, out);
}
